// Attention_34454227648722
// MI455X (gfx1250) — matrix operand order repaired and checked by reading, not compiled, not run on hardware
//
#include <hip/hip_runtime.h>

// ---------------------------------------------------------------------------
// Types for CDNA5 WMMA
// ---------------------------------------------------------------------------
typedef float v8f  __attribute__((ext_vector_type(8)));
typedef __bf16 v16bf __attribute__((ext_vector_type(16)));

union FragBF {
  v16bf bf;
  uint4 q[2];
  unsigned short us[16];
};

__device__ __forceinline__ unsigned short f2bf(float f) {
  union { float f; unsigned int u; } v;
  v.f = f;
  unsigned int u = v.u;
  u += 0x7fffu + ((u >> 16) & 1u);   // round-to-nearest-even
  return (unsigned short)(u >> 16);
}

// A-fragment (16x32 bf16): lane = row M, elements 0..7 -> K = half*8+e,
// elements 8..15 -> K = 16 + half*8 + e   (ISA 7.12.2, 16-bit A 16x32)
__device__ __forceinline__ FragBF load_frag_a(const unsigned short* rowK0, int half) {
  FragBF f;
  f.q[0] = *(const uint4*)(rowK0 + half * 8);
  f.q[1] = *(const uint4*)(rowK0 + 16 + half * 8);
  return f;
}

// B-fragment (32x16 bf16): lane = column N, elements 0..7 -> K = half*8+e,
// elements 8..15 -> K = 16 + half*8 + e   (same k order as A)
__device__ __forceinline__ FragBF load_frag_b(const unsigned short* colK0, int half) {
  FragBF f;
  f.q[0] = *(const uint4*)(colK0 + half * 8);
  f.q[1] = *(const uint4*)(colK0 + 16 + half * 8);
  return f;
}

__device__ __forceinline__ v8f wmma_bf16(const FragBF& a, const FragBF& b, v8f c) {
  return __builtin_amdgcn_wmma_f32_16x16x32_bf16(
      /*neg_a=*/false, a.bf, /*neg_b=*/false, b.bf,
      /*c_mod=*/(short)0, c, /*reuse_a=*/false, /*reuse_b=*/false);
}

// ---------------------------------------------------------------------------
// Problem constants (ViT-Base attention)
// ---------------------------------------------------------------------------
#define BB   64
#define NN   197
#define HH   12
#define DHD  64
#define DD   768
#define QKVD 2304
#define NT   13      // ceil(197/16) key/query tiles
#define NPAD 208     // NT*16
#define ROWS (BB * NN)   // 12608 (multiple of 16)

// ---------------------------------------------------------------------------
// Conversion kernels
// ---------------------------------------------------------------------------
__global__ void k_cvt_x(const float* __restrict__ x, unsigned short* __restrict__ xb,
                        int n, float* __restrict__ kld) {
  int i = blockIdx.x * blockDim.x + threadIdx.x;
  if (i == 0) *kld = 0.0f;
  if (i < n) xb[i] = f2bf(x[i]);
}

// w: [K,N] row-major fp32  ->  wt: [N,K] bf16 (transposed)
__global__ void k_cvt_wT(const float* __restrict__ w, unsigned short* __restrict__ wt,
                         int K, int N) {
  int i = blockIdx.x * blockDim.x + threadIdx.x;
  if (i < K * N) {
    int n = i / K;
    int k = i - n * K;
    wt[(size_t)n * K + k] = f2bf(w[(size_t)k * N + n]);
  }
}

// ---------------------------------------------------------------------------
// bf16 WMMA GEMM:  C[M,N] = A[M,K] * B[K,N]   (Bt stored transposed [N,K])
// One 16x64 output tile per wave: 1 A-fragment reused across 4 B-fragments /
// 4 accumulators. Explicit software pipeline: next K-step's fragments are
// loaded into distinct registers before the current step's 4 WMMAs issue.
// ---------------------------------------------------------------------------
__global__ __launch_bounds__(128) void k_gemm_bf16(
    const unsigned short* __restrict__ A,   // [M,K] bf16
    const unsigned short* __restrict__ Bt,  // [N,K] bf16
    unsigned short* __restrict__ Cb,        // bf16 out (or unused)
    float* __restrict__ Cf,                 // f32 out (or null)
    const float* __restrict__ bias,         // f32 bias (or null)
    int M, int N, int K) {
  const int lane = threadIdx.x & 31;
  const int wave = threadIdx.x >> 5;
  const int half = lane >> 4;
  const int l16  = lane & 15;

  const int nt64 = N >> 6;                 // 64-wide column groups
  int tile = blockIdx.x * 4 + wave;
  int mt = tile / nt64;
  int nb = tile - mt * nt64;
  if (mt >= (M >> 4)) return;              // wave-uniform

  const unsigned short* arow = A  + (size_t)(mt * 16 + l16) * K;
  const unsigned short* bc0  = Bt + (size_t)(nb * 64 +  0 + l16) * K;
  const unsigned short* bc1  = Bt + (size_t)(nb * 64 + 16 + l16) * K;
  const unsigned short* bc2  = Bt + (size_t)(nb * 64 + 32 + l16) * K;
  const unsigned short* bc3  = Bt + (size_t)(nb * 64 + 48 + l16) * K;

  // prologue: fragments for k0 = 0
  FragBF af = load_frag_a(arow, half);
  FragBF b0 = load_frag_b(bc0, half);
  FragBF b1 = load_frag_b(bc1, half);
  FragBF b2 = load_frag_b(bc2, half);
  FragBF b3 = load_frag_b(bc3, half);

  v8f c0 = {}, c1 = {}, c2 = {}, c3 = {};
  for (int k0 = 0; k0 < K; k0 += 32) {
    // uniform clamp: last iteration re-loads current step (discarded)
    int kn = (k0 + 32 < K) ? (k0 + 32) : k0;
    __builtin_prefetch(arow + k0 + 128, 0, 3);   // streaming operand only
    FragBF afn = load_frag_a(arow + kn, half);
    FragBF b0n = load_frag_b(bc0 + kn, half);
    FragBF b1n = load_frag_b(bc1 + kn, half);
    FragBF b2n = load_frag_b(bc2 + kn, half);
    FragBF b3n = load_frag_b(bc3 + kn, half);

    c0 = wmma_bf16(af, b0, c0);
    c1 = wmma_bf16(af, b1, c1);
    c2 = wmma_bf16(af, b2, c2);
    c3 = wmma_bf16(af, b3, c3);

    af = afn; b0 = b0n; b1 = b1n; b2 = b2n; b3 = b3n;
  }

  if (Cf) {
#pragma unroll
    for (int t = 0; t < 4; ++t) {
      const v8f& acc = (t == 0) ? c0 : (t == 1) ? c1 : (t == 2) ? c2 : c3;
      int col = nb * 64 + t * 16 + l16;
      float bv = bias ? bias[col] : 0.0f;
#pragma unroll
      for (int r = 0; r < 8; ++r) {
        int row = mt * 16 + r + half * 8;
        Cf[(size_t)row * N + col] = acc[r] + bv;
      }
    }
  } else {
#pragma unroll
    for (int t = 0; t < 4; ++t) {
      const v8f& acc = (t == 0) ? c0 : (t == 1) ? c1 : (t == 2) ? c2 : c3;
      int col = nb * 64 + t * 16 + l16;
#pragma unroll
      for (int r = 0; r < 8; ++r) {
        int row = mt * 16 + r + half * 8;
        Cb[(size_t)row * N + col] = f2bf(acc[r]);
      }
    }
  }
}

// ---------------------------------------------------------------------------
// Fused attention: one block per (b,h). 4 waves; wave w owns query tiles w, w+4, ... (16 rows each).
// qkv: [B*N, 2304] bf16 (q | k | v, head-major inside each 768-chunk)
// ao : [B*N, 768]  bf16 attention output (pre out-projection)
// ---------------------------------------------------------------------------
__global__ __launch_bounds__(128) void k_attention(
    const unsigned short* __restrict__ qkv,
    unsigned short* __restrict__ ao,
    float* __restrict__ kld_out) {
  __shared__ unsigned short vT[DHD][NPAD];      // V transposed: [dh][key]
  __shared__ unsigned short pS[4][16 * 32];     // per-wave P staging tile

  const int b    = blockIdx.x / HH;
  const int h    = blockIdx.x - b * HH;
  const int tid  = threadIdx.x;
  const int lane = tid & 31;
  const int wave = tid >> 5;
  const int half = lane >> 4;
  const int l16  = lane & 15;

  const unsigned short* qbase = qkv + (size_t)b * NN * QKVD + h * DHD;
  const unsigned short* kbase = qbase + 768;
  const unsigned short* vbase = qbase + 1536;

  // Stage V transposed into LDS (zero-pad keys 197..207)
  for (int i = tid; i < NPAD * DHD; i += 128) {
    int key = i >> 6;
    int dh  = i & 63;
    unsigned short v = 0;
    if (key < NN) v = vbase[(size_t)key * QKVD + dh];
    vT[dh][key] = v;
  }
  __syncthreads();

  const float scale = 0.125f;   // 64^-0.5
  float kacc = 0.0f;

  for (int qt = wave; qt < NT; qt += 4) {
    // ---- Q fragments (two K-steps over DH=64) ----
    int qrow = qt * 16 + l16;
    int qrc  = qrow < NN ? qrow : NN - 1;          // clamp (rows masked later)
    const unsigned short* qp = qbase + (size_t)qrc * QKVD;
    FragBF qf0 = load_frag_a(qp, half);            // dh 0..31
    FragBF qf1 = load_frag_a(qp + 32, half);       // dh 32..63

    // ---- S = scale * Q K^T for all 13 key tiles ----
    v8f s[NT];
#pragma unroll
    for (int j = 0; j < NT; ++j) {
      int keyc = j * 16 + l16;
      int keyl = keyc < NN ? keyc : NN - 1;        // clamp loads, keep EXEC full
      const unsigned short* kp = kbase + (size_t)keyl * QKVD;
      FragBF kf0 = load_frag_b(kp, half);
      FragBF kf1 = load_frag_b(kp + 32, half);
      v8f c = {};
      c = wmma_bf16(qf0, kf0, c);
      c = wmma_bf16(qf1, kf1, c);
      bool vcol = keyc < NN;
#pragma unroll
      for (int r = 0; r < 8; ++r) s[j][r] = vcol ? c[r] * scale : -1e30f;
    }

    // ---- row max (row lives across lanes 0..15 of each half) ----
    v8f mrow;
#pragma unroll
    for (int r = 0; r < 8; ++r) {
      float m = -1e30f;
#pragma unroll
      for (int j = 0; j < NT; ++j) m = fmaxf(m, s[j][r]);
      mrow[r] = m;
    }
#pragma unroll
    for (int off = 1; off < 16; off <<= 1) {
#pragma unroll
      for (int r = 0; r < 8; ++r)
        mrow[r] = fmaxf(mrow[r], __shfl_xor(mrow[r], off, 32));
    }

    // ---- exp + row sum ----
    v8f lsum = {};
#pragma unroll
    for (int j = 0; j < NT; ++j) {
#pragma unroll
      for (int r = 0; r < 8; ++r) {
        float p = __expf(s[j][r] - mrow[r]);
        s[j][r] = p;
        lsum[r] += p;
      }
    }
#pragma unroll
    for (int off = 1; off < 16; off <<= 1) {
#pragma unroll
      for (int r = 0; r < 8; ++r) lsum[r] += __shfl_xor(lsum[r], off, 32);
    }
    v8f invl;
#pragma unroll
    for (int r = 0; r < 8; ++r) invl[r] = 1.0f / lsum[r];

    // ---- KLD partial: sum_j log(attn*65 + 1e-12) over valid (row,col) ----
#pragma unroll
    for (int j = 0; j < NT; ++j) {
      bool vcol = (j * 16 + l16) < NN;
#pragma unroll
      for (int r = 0; r < 8; ++r) {
        int row = qt * 16 + r + half * 8;
        if (vcol && row < NN) {
          float a = s[j][r] * invl[r];
          kacc += __logf(a * 65.0f + 1e-12f);
        }
      }
    }

    // ---- O = P V  (process key tiles in pairs -> K=32 WMMA steps) ----
    v8f o0 = {}, o1 = {}, o2 = {}, o3 = {};
    unsigned short* myP = pS[wave];
#pragma unroll
    for (int jj = 0; jj < 7; ++jj) {
      int j0 = jj * 2, j1 = jj * 2 + 1;
      // D-layout -> A-layout transpose through per-wave LDS tile
#pragma unroll
      for (int r = 0; r < 8; ++r) {
        int Mr = r + half * 8;
        myP[Mr * 32 + l16]      = f2bf(s[j0][r]);
        myP[Mr * 32 + 16 + l16] = (j1 < NT) ? f2bf(s[j1][r]) : (unsigned short)0;
      }
      asm volatile("s_wait_dscnt 0" ::: "memory");
      FragBF pf = load_frag_a(myP + l16 * 32, half);

      FragBF vf0 = load_frag_b(&vT[ 0 + l16][jj * 32], half);
      FragBF vf1 = load_frag_b(&vT[16 + l16][jj * 32], half);
      FragBF vf2 = load_frag_b(&vT[32 + l16][jj * 32], half);
      FragBF vf3 = load_frag_b(&vT[48 + l16][jj * 32], half);
      o0 = wmma_bf16(pf, vf0, o0);
      o1 = wmma_bf16(pf, vf1, o1);
      o2 = wmma_bf16(pf, vf2, o2);
      o3 = wmma_bf16(pf, vf3, o3);
    }

    // ---- normalize rows by 1/l and store bf16 ----
#pragma unroll
    for (int r = 0; r < 8; ++r) {
      int row = qt * 16 + r + half * 8;
      if (row < NN) {
        size_t o = ((size_t)b * NN + row) * DD + h * DHD;
        ao[o +  0 + l16] = f2bf(o0[r] * invl[r]);
        ao[o + 16 + l16] = f2bf(o1[r] * invl[r]);
        ao[o + 32 + l16] = f2bf(o2[r] * invl[r]);
        ao[o + 48 + l16] = f2bf(o3[r] * invl[r]);
      }
    }
  }

  // ---- wave-wide KLD reduction, one atomic per wave ----
#pragma unroll
  for (int off = 1; off < 32; off <<= 1) kacc += __shfl_xor(kacc, off, 32);
  if (lane == 0) {
    const float norm = (-1.0f / 65.0f) / (float)(BB * HH * NN);
    atomicAdd(kld_out, kacc * norm);
  }
}

// ---------------------------------------------------------------------------
// Host-side launcher
// ---------------------------------------------------------------------------
extern "C" void kernel_launch(void* const* d_in, const int* in_sizes, int n_in,
                              void* d_out, int out_size, void* d_ws, size_t ws_size,
                              hipStream_t stream) {
  (void)in_sizes; (void)n_in; (void)out_size; (void)ws_size;

  const float* x     = (const float*)d_in[0];   // [64,197,768]
  const float* w_qkv = (const float*)d_in[1];   // [768,2304]
  const float* w_out = (const float*)d_in[2];   // [768,768]
  const float* b_out = (const float*)d_in[3];   // [768]

  float* out = (float*)d_out;                         // [64,197,768]
  float* kld = out + (size_t)ROWS * DD;               // scalar at the end

  // workspace layout (all offsets 256B aligned by construction)
  char* ws = (char*)d_ws;
  unsigned short* xb    = (unsigned short*)(ws);                         // 12608*768  bf16
  unsigned short* wqkvT = (unsigned short*)(ws + 19365888ULL);           // 2304*768   bf16
  unsigned short* woutT = (unsigned short*)(ws + 19365888ULL + 3538944); // 768*768    bf16
  unsigned short* qkvb  = (unsigned short*)(ws + 24084480ULL);           // 12608*2304 bf16
  unsigned short* aob   = (unsigned short*)(ws + 82182144ULL);           // 12608*768  bf16

  const int nX = ROWS * DD;   // 9,682,944
  k_cvt_x<<<(nX + 255) / 256, 256, 0, stream>>>(x, xb, nX, kld);
  k_cvt_wT<<<(DD * QKVD + 255) / 256, 256, 0, stream>>>(w_qkv, wqkvT, DD, QKVD);
  k_cvt_wT<<<(DD * DD + 255) / 256, 256, 0, stream>>>(w_out, woutT, DD, DD);

  // QKV projection: [12608,768] x [768,2304] -> bf16 qkv
  {
    int tiles = (ROWS / 16) * (QKVD / 64);  // 788*36 = 28368
    k_gemm_bf16<<<tiles / 4, 128, 0, stream>>>(xb, wqkvT, qkvb, nullptr, nullptr,
                                               ROWS, QKVD, DD);
  }

  // Fused attention + KLD
  k_attention<<<BB * HH, 128, 0, stream>>>(qkvb, aob, kld);

  // Output projection: [12608,768] x [768,768] + bias -> f32 out
  {
    int tiles = (ROWS / 16) * (DD / 64);    // 788*12 = 9456
    k_gemm_bf16<<<tiles / 4, 128, 0, stream>>>(aob, woutT, nullptr, out, b_out,
                                               ROWS, DD, DD);
  }
}
